// DMNPlus_81801947119949
// MI455X (gfx1250) — compile-verified, matrix operand order checked by reading, not run on hardware
//
#include <hip/hip_runtime.h>
#include <cstdint>
#include <cstddef>

// ---------------------------------------------------------------------------
// DMN+ forward on gfx1250 (MI455X).  All GEMMs use v_wmma_f32_16x16x32_f16.
// GEMM tiles are staged with the CDNA5 async global->LDS engine
// (global_load_async_to_lds_b128 + s_wait_asynccnt) in a ping-pong pipeline.
// Recurrent scans (bi-GRU, question GRU, AGRU) run inside single persistent
// workgroups with the hidden state LDS-resident (wave32, WGP 320KB LDS).
// ---------------------------------------------------------------------------

typedef __attribute__((ext_vector_type(16))) _Float16 v16h;
typedef __attribute__((ext_vector_type(8)))  float    v8f;

#define HD   512
#define BN   64
#define SN   64
#define TN   32
#define TQN  32
#define VOC  32000
#define H3   1536
#define H4   2048

__device__ __forceinline__ v8f zero8() {
  v8f z;
#pragma unroll
  for (int i = 0; i < 8; ++i) z[i] = 0.0f;
  return z;
}

__device__ __forceinline__ v8f wmma32(v16h a, v16h b, v8f c) {
  // D = A(16x32 f16) * B(32x16 f16) + C(16x16 f32)
  return __builtin_amdgcn_wmma_f32_16x16x32_f16(
      /*neg_a=*/false, a, /*neg_b=*/false, b,
      /*c_mod=*/(short)0, c, /*reuse_a=*/false, /*reuse_b=*/false);
}

// CDNA5 async copy: 16 bytes global -> LDS, tracked by ASYNCcnt.
// lds_off is the workgroup-relative LDS byte address (low 32 bits of the
// generic pointer to a __shared__ object).
__device__ __forceinline__ void async_copy_b128(unsigned lds_off,
                                                const void* gptr) {
  asm volatile("global_load_async_to_lds_b128 %0, %1, off"
               :: "v"(lds_off), "v"(gptr)
               : "memory");
}

__device__ __forceinline__ void wait_async0() {
  asm volatile("s_wait_asynccnt 0" ::: "memory");
}

// Load one 16x32 f16 fragment (A-layout; B uses the symmetric N-row layout)
// from a row-major matrix with leading dimension ld (in elements).
// lane 0..15  -> row row0+lane,    K in {k0..k0+7} U {k0+16..k0+23}
// lane 16..31 -> row row0+lane-16, K in {k0+8..k0+15} U {k0+24..k0+31}
__device__ __forceinline__ v16h load_frag(const _Float16* __restrict__ p,
                                          int ld, int row0, int k0) {
  const int lane = threadIdx.x & 31;
  const int r    = row0 + (lane & 15);
  const int kk   = k0 + ((lane & 16) >> 1);   // +8 for upper half-wave
  const _Float16* q = p + (size_t)r * (size_t)ld + kk;
  v16h f;
#pragma unroll
  for (int i = 0; i < 8; ++i) { f[i] = q[i]; f[i + 8] = q[i + 16]; }
  return f;
}

__device__ __forceinline__ float sigmoidf_(float x) {
  return 1.0f / (1.0f + expf(-x));
}

// ---------------------------------------------------------------------------
// Weight conversion kernels
// ---------------------------------------------------------------------------
__global__ void f2h_kernel(const float* __restrict__ in,
                           _Float16* __restrict__ out, int n) {
  int i = blockIdx.x * blockDim.x + threadIdx.x;
  if (i < n) out[i] = (_Float16)in[i];
}

// out[n*512+k] = in[k*512+n]   (for 512x512 matrices used un-transposed)
__global__ void t512_f2h_kernel(const float* __restrict__ in,
                                _Float16* __restrict__ out) {
  int n = blockIdx.x * 16 + threadIdx.x;
  int k = blockIdx.y * 16 + threadIdx.y;
  out[n * HD + k] = (_Float16)in[k * HD + n];
}

// ---------------------------------------------------------------------------
// Embedding kernels (emb row 0 forced to zero, per reference)
// ---------------------------------------------------------------------------
__global__ void embed_facts_kernel(const int* __restrict__ ctx,
                                   const float* __restrict__ emb,
                                   _Float16* __restrict__ facts_in16) {
  const int row = blockIdx.x;            // b*64 + s, 0..4095
  const int s   = row & 63;
  for (int h = threadIdx.x; h < HD; h += blockDim.x) {
    float acc = 0.0f;
    for (int t = 0; t < TN; ++t) {
      int idx = ctx[row * TN + t];
      if (idx != 0) acc += emb[(size_t)idx * HD + h];
    }
    float fs = (float)s, fe = (float)h;
    float pe = 1.0f - fs / (float)SN - (fe / (float)HD) * (1.0f - 2.0f * fs / (float)SN);
    facts_in16[(size_t)row * HD + h] = (_Float16)(acc * pe);
  }
}

__global__ void embed_q_kernel(const int* __restrict__ ques,
                               const float* __restrict__ emb,
                               _Float16* __restrict__ qemb16) {
  const int row = blockIdx.x;            // b*32 + t, 0..2047
  const int idx = ques[row];
  for (int h = threadIdx.x; h < HD; h += blockDim.x) {
    float v = (idx != 0) ? emb[(size_t)idx * HD + h] : 0.0f;
    qemb16[(size_t)row * HD + h] = (_Float16)v;
  }
}

// ---------------------------------------------------------------------------
// Generic WMMA GEMM:  C(MxN,f32) = act( A(MxK,f16) * W(NxK,f16)^T + bias )
// M,N multiples of 64, K multiple of 32.  256 threads / block, 64x64 tile.
// Tiles staged via async global->LDS copies, double buffered (one barrier
// per K-step).  act: 0=none 1=tanh 2=relu
// ---------------------------------------------------------------------------
__global__ __launch_bounds__(256)
void wmma_gemm_kernel(const _Float16* __restrict__ A,
                      const _Float16* __restrict__ W,
                      const float* __restrict__ bias,
                      float* __restrict__ C,
                      int M, int N, int K, int act) {
  __shared__ _Float16 As[2][64 * 40];
  __shared__ _Float16 Ws[2][64 * 40];
  const int m0 = blockIdx.y * 64, n0 = blockIdx.x * 64;
  const int tid  = threadIdx.x;
  const int wave = tid >> 5, lane = tid & 31;
  const int ct  = wave & 3;            // column tile 0..3
  const int rt0 = (wave >> 2) * 2;     // row tiles {rt0, rt0+1}
  v8f acc0 = zero8(), acc1 = zero8();

  const int lrow = tid >> 2;           // 0..63
  const int lcol = (tid & 3) * 8;      // 0,8,16,24  (16B-aligned chunks)
  const _Float16* ag = A + (size_t)(m0 + lrow) * K + lcol;
  const _Float16* wg = W + (size_t)(n0 + lrow) * K + lcol;

  unsigned aoff[2], woff[2];
  aoff[0] = (unsigned)(uintptr_t)(&As[0][lrow * 40 + lcol]);
  aoff[1] = (unsigned)(uintptr_t)(&As[1][lrow * 40 + lcol]);
  woff[0] = (unsigned)(uintptr_t)(&Ws[0][lrow * 40 + lcol]);
  woff[1] = (unsigned)(uintptr_t)(&Ws[1][lrow * 40 + lcol]);

  // prime pipeline: fill buffer 0
  async_copy_b128(aoff[0], ag);
  async_copy_b128(woff[0], wg);

  const int nk = K >> 5;
  for (int i = 0; i < nk; ++i) {
    const int cur = i & 1, nxt = cur ^ 1;
    wait_async0();       // this wave's async writes to buf[cur] have landed
    __syncthreads();     // => every wave's writes landed; prev reads retired
    if (i + 1 < nk) {    // refill the other buffer while we compute
      async_copy_b128(aoff[nxt], ag + (size_t)(i + 1) * 32);
      async_copy_b128(woff[nxt], wg + (size_t)(i + 1) * 32);
    }
    v16h bf = load_frag(Ws[cur], 40, ct * 16, 0);
    v16h a0 = load_frag(As[cur], 40, rt0 * 16, 0);
    v16h a1 = load_frag(As[cur], 40, rt0 * 16 + 16, 0);
    acc0 = wmma32(a0, bf, acc0);
    acc1 = wmma32(a1, bf, acc1);
  }

  const int cn  = n0 + ct * 16 + (lane & 15);
  const int hi8 = (lane & 16) >> 1;
  const float bv = bias ? bias[cn] : 0.0f;
#pragma unroll
  for (int e = 0; e < 8; ++e) {
    float v0 = acc0[e] + bv;
    float v1 = acc1[e] + bv;
    if (act == 1) { v0 = tanhf(v0); v1 = tanhf(v1); }
    else if (act == 2) { v0 = fmaxf(v0, 0.0f); v1 = fmaxf(v1, 0.0f); }
    int rm0 = m0 + rt0 * 16 + e + hi8;
    int rm1 = m0 + (rt0 + 1) * 16 + e + hi8;
    C[(size_t)rm0 * N + cn] = v0;
    C[(size_t)rm1 * N + cn] = v1;
  }
}

// ---------------------------------------------------------------------------
// Fused GRU scans.  grid = 3 blocks x 1024 threads (32 waves):
//   block 0: forward facts GRU  (64 steps, writes hs_f)
//   block 1: backward facts GRU (64 steps, writes hs_b)
//   block 2: question GRU       (32 steps, writes final q)
// Hidden state (64x512) resides in LDS: f16 copy for WMMA A-fragments,
// f32 master copy for the update.  Wave w owns hidden columns [16w,16w+16)
// and computes r/z/n gate columns for all 64 batch rows (12 WMMA tiles).
// Dynamic LDS = 64*520*2 + 64*512*4 = 197632 bytes.
// ---------------------------------------------------------------------------
#define SCAN_LDS (64 * 520 * 2 + 64 * 512 * 4)

__global__ __launch_bounds__(1024)
void gru_scan_kernel(const float* __restrict__ xg_f,
                     const float* __restrict__ xg_b,
                     const float* __restrict__ xg_q,
                     const _Float16* __restrict__ whh_f,
                     const _Float16* __restrict__ whh_b,
                     const _Float16* __restrict__ whh_q,
                     const float* __restrict__ bhh_f,
                     const float* __restrict__ bhh_b,
                     const float* __restrict__ bhh_q,
                     float* __restrict__ hs_f,
                     float* __restrict__ hs_b,
                     float* __restrict__ q32,
                     _Float16* __restrict__ q16) {
  extern __shared__ char smem[];
  _Float16* h16 = (_Float16*)smem;                       // 64 x (stride 520)
  float*    h32 = (float*)(smem + 64 * 520 * sizeof(_Float16)); // 64 x 512

  const float* xg; const _Float16* whh; const float* bhh;
  float* hs_out = nullptr; int steps; bool rev = false;
  if (blockIdx.x == 0)      { xg = xg_f; whh = whh_f; bhh = bhh_f; hs_out = hs_f; steps = SN; }
  else if (blockIdx.x == 1) { xg = xg_b; whh = whh_b; bhh = bhh_b; hs_out = hs_b; steps = SN; rev = true; }
  else                      { xg = xg_q; whh = whh_q; bhh = bhh_q; steps = TQN; }

  const int tid = threadIdx.x, wave = tid >> 5, lane = tid & 31;
  const int cb  = wave * 16;
  const int col = cb + (lane & 15);
  const int hi8 = (lane & 16) >> 1;

  for (int i = tid; i < 64 * 520; i += 1024) h16[i] = (_Float16)0.0f;
  for (int i = tid; i < 64 * 512; i += 1024) h32[i] = 0.0f;
  __syncthreads();

  const float b0 = bhh[col], b1 = bhh[HD + col], b2 = bhh[2 * HD + col];

  for (int it = 0; it < steps; ++it) {
    const int s = rev ? (steps - 1 - it) : it;

    v8f acc[12];                         // [gate*4 + row_tile]
#pragma unroll
    for (int i = 0; i < 12; ++i) acc[i] = zero8();

    for (int kb = 0; kb < HD; kb += 32) {
      v16h a0 = load_frag(h16, 520, 0,  kb);
      v16h a1 = load_frag(h16, 520, 16, kb);
      v16h a2 = load_frag(h16, 520, 32, kb);
      v16h a3 = load_frag(h16, 520, 48, kb);
      v16h w0 = load_frag(whh, HD, cb,            kb);  // r gate
      v16h w1 = load_frag(whh, HD, HD + cb,       kb);  // z gate
      v16h w2 = load_frag(whh, HD, 2 * HD + cb,   kb);  // n gate
      acc[0] = wmma32(a0, w0, acc[0]);  acc[1] = wmma32(a1, w0, acc[1]);
      acc[2] = wmma32(a2, w0, acc[2]);  acc[3] = wmma32(a3, w0, acc[3]);
      acc[4] = wmma32(a0, w1, acc[4]);  acc[5] = wmma32(a1, w1, acc[5]);
      acc[6] = wmma32(a2, w1, acc[6]);  acc[7] = wmma32(a3, w1, acc[7]);
      acc[8] = wmma32(a0, w2, acc[8]);  acc[9] = wmma32(a1, w2, acc[9]);
      acc[10] = wmma32(a2, w2, acc[10]); acc[11] = wmma32(a3, w2, acc[11]);
    }

    float hnew[32];
#pragma unroll
    for (int rt = 0; rt < 4; ++rt) {
#pragma unroll
      for (int e = 0; e < 8; ++e) {
        const int m = rt * 16 + e + hi8;              // batch row
        const size_t xb = ((size_t)m * steps + s) * H3;
        float gr = xg[xb + col]            + acc[0 + rt][e] + b0;
        float gz = xg[xb + HD + col]       + acc[4 + rt][e] + b1;
        float gn = acc[8 + rt][e] + b2;
        float r  = sigmoidf_(gr);
        float z  = sigmoidf_(gz);
        float n  = tanhf(xg[xb + 2 * HD + col] + r * gn);
        float ho = h32[m * HD + col];
        hnew[rt * 8 + e] = (1.0f - z) * n + z * ho;
      }
    }
    __syncthreads();                     // all reads of h done
#pragma unroll
    for (int rt = 0; rt < 4; ++rt) {
#pragma unroll
      for (int e = 0; e < 8; ++e) {
        const int m = rt * 16 + e + hi8;
        const float v = hnew[rt * 8 + e];
        h32[m * HD + col] = v;
        h16[m * 520 + col] = (_Float16)v;
        if (hs_out) hs_out[((size_t)m * steps + s) * HD + col] = v;
      }
    }
    __syncthreads();                     // writes visible before next step
  }

  if (!hs_out) {                         // question GRU: emit final hidden
    for (int i = tid; i < 64 * 512; i += 1024) {
      float v = h32[i];
      q32[i] = v;
      q16[i] = (_Float16)v;
    }
  }
}

// ---------------------------------------------------------------------------
// AGRU scan (attention GRU).  1 block x 1024 threads, c-state in LDS.
//   r  = sigmoid(A1[b,s,:] + c @ Ur)      (A1 = facts@Wr + br, precomputed)
//   ht = tanh   (A2[b,s,:] + r * (c @ U)) (A2 = facts@W  + br, precomputed)
//   c  = g*ht + (1-g)*c
// ---------------------------------------------------------------------------
__global__ __launch_bounds__(1024)
void agru_scan_kernel(const float* __restrict__ A1,
                      const float* __restrict__ A2,
                      const _Float16* __restrict__ urT,
                      const _Float16* __restrict__ uT,
                      const float* __restrict__ G,
                      float* __restrict__ Cout) {
  extern __shared__ char smem[];
  _Float16* c16 = (_Float16*)smem;
  float*    c32 = (float*)(smem + 64 * 520 * sizeof(_Float16));

  const int tid = threadIdx.x, wave = tid >> 5, lane = tid & 31;
  const int cb  = wave * 16;
  const int col = cb + (lane & 15);
  const int hi8 = (lane & 16) >> 1;

  for (int i = tid; i < 64 * 520; i += 1024) c16[i] = (_Float16)0.0f;
  for (int i = tid; i < 64 * 512; i += 1024) c32[i] = 0.0f;
  __syncthreads();

  for (int s = 0; s < SN; ++s) {
    v8f acc[8];                          // [mat*4 + row_tile]  mat0=Ur mat1=U
#pragma unroll
    for (int i = 0; i < 8; ++i) acc[i] = zero8();

    for (int kb = 0; kb < HD; kb += 32) {
      v16h a0 = load_frag(c16, 520, 0,  kb);
      v16h a1 = load_frag(c16, 520, 16, kb);
      v16h a2 = load_frag(c16, 520, 32, kb);
      v16h a3 = load_frag(c16, 520, 48, kb);
      v16h w0 = load_frag(urT, HD, cb, kb);
      v16h w1 = load_frag(uT,  HD, cb, kb);
      acc[0] = wmma32(a0, w0, acc[0]);  acc[1] = wmma32(a1, w0, acc[1]);
      acc[2] = wmma32(a2, w0, acc[2]);  acc[3] = wmma32(a3, w0, acc[3]);
      acc[4] = wmma32(a0, w1, acc[4]);  acc[5] = wmma32(a1, w1, acc[5]);
      acc[6] = wmma32(a2, w1, acc[6]);  acc[7] = wmma32(a3, w1, acc[7]);
    }

    float cnew[32];
#pragma unroll
    for (int rt = 0; rt < 4; ++rt) {
#pragma unroll
      for (int e = 0; e < 8; ++e) {
        const int m = rt * 16 + e + hi8;
        const size_t idx = ((size_t)m * SN + s) * HD + col;
        float r  = sigmoidf_(A1[idx] + acc[0 + rt][e]);
        float ht = tanhf(A2[idx] + r * acc[4 + rt][e]);
        float g  = G[m * SN + s];
        cnew[rt * 8 + e] = g * ht + (1.0f - g) * c32[m * HD + col];
      }
    }
    __syncthreads();
#pragma unroll
    for (int rt = 0; rt < 4; ++rt) {
#pragma unroll
      for (int e = 0; e < 8; ++e) {
        const int m = rt * 16 + e + hi8;
        const float v = cnew[rt * 8 + e];
        c32[m * HD + col] = v;
        c16[m * 520 + col] = (_Float16)v;
      }
    }
    __syncthreads();
  }

  for (int i = tid; i < 64 * 512; i += 1024) Cout[i] = c32[i];
}

// ---------------------------------------------------------------------------
// Elementwise helpers
// ---------------------------------------------------------------------------
__global__ void facts_add_kernel(const float* __restrict__ hf,
                                 const float* __restrict__ hb,
                                 float* __restrict__ f32o,
                                 _Float16* __restrict__ f16o, int n) {
  int i = blockIdx.x * blockDim.x + threadIdx.x;
  if (i < n) { float v = hf[i] + hb[i]; f32o[i] = v; f16o[i] = (_Float16)v; }
}

__global__ void build_z_kernel(const float* __restrict__ facts,
                               const float* __restrict__ q,
                               const float* __restrict__ M,
                               _Float16* __restrict__ z) {
  const int row = blockIdx.x;            // b*64 + s
  const int b   = row >> 6;
  for (int h = threadIdx.x; h < HD; h += blockDim.x) {
    float f = facts[(size_t)row * HD + h];
    float qv = q[b * HD + h];
    float mv = M[b * HD + h];
    size_t base = (size_t)row * H4;
    z[base + h]            = (_Float16)(f * qv);
    z[base + HD + h]       = (_Float16)(f * mv);
    z[base + 2 * HD + h]   = (_Float16)fabsf(f - qv);
    z[base + 3 * HD + h]   = (_Float16)fabsf(f - mv);
  }
}

__global__ void attn_kernel(const float* __restrict__ g1,
                            const float* __restrict__ z2w,
                            const float* __restrict__ z2b,
                            float* __restrict__ G) {
  const int b = blockIdx.x, s = threadIdx.x;   // 64 threads
  const float* row = g1 + ((size_t)b * SN + s) * HD;
  float acc = z2b[0];
  for (int k = 0; k < HD; ++k) acc += row[k] * z2w[k];
  __shared__ float lg[64];
  __shared__ float le[64];
  lg[s] = acc; __syncthreads();
  float mx = lg[0];
  for (int i = 1; i < SN; ++i) mx = fmaxf(mx, lg[i]);
  float ex = expf(acc - mx);
  le[s] = ex; __syncthreads();
  float sum = 0.0f;
  for (int i = 0; i < SN; ++i) sum += le[i];
  G[b * SN + s] = ex / sum;
}

__global__ void cat3_kernel(const float* __restrict__ M,
                            const float* __restrict__ C,
                            const float* __restrict__ q,
                            _Float16* __restrict__ cat) {
  const int b = blockIdx.x;
  for (int j = threadIdx.x; j < H3; j += blockDim.x) {
    float v = (j < HD) ? M[b * HD + j]
            : (j < 2 * HD) ? C[b * HD + (j - HD)]
            : q[b * HD + (j - 2 * HD)];
    cat[(size_t)b * H3 + j] = (_Float16)v;
  }
}

__global__ void cat2_kernel(const float* __restrict__ M,
                            const float* __restrict__ q,
                            _Float16* __restrict__ cat) {
  const int b = blockIdx.x;
  for (int j = threadIdx.x; j < 2 * HD; j += blockDim.x) {
    float v = (j < HD) ? M[b * HD + j] : q[b * HD + (j - HD)];
    cat[(size_t)b * 2 * HD + j] = (_Float16)v;
  }
}

__global__ void copyf_kernel(const float* __restrict__ in,
                             float* __restrict__ out, int n) {
  int i = blockIdx.x * blockDim.x + threadIdx.x;
  if (i < n) out[i] = in[i];
}

// ---------------------------------------------------------------------------
// Host launcher
// ---------------------------------------------------------------------------
extern "C" void kernel_launch(void* const* d_in, const int* in_sizes, int n_in,
                              void* d_out, int out_size, void* d_ws, size_t ws_size,
                              hipStream_t stream) {
  (void)in_sizes; (void)n_in; (void)out_size; (void)ws_size;
  const int*   ctx   = (const int*)  d_in[0];
  const int*   ques  = (const int*)  d_in[1];
  const float* emb   = (const float*)d_in[2];
  const float* Wih_f = (const float*)d_in[3];
  const float* Whh_f = (const float*)d_in[4];
  const float* bih_f = (const float*)d_in[5];
  const float* bhh_f = (const float*)d_in[6];
  const float* Wih_b = (const float*)d_in[7];
  const float* Whh_b = (const float*)d_in[8];
  const float* bih_b = (const float*)d_in[9];
  const float* bhh_b = (const float*)d_in[10];
  const float* qWih  = (const float*)d_in[11];
  const float* qWhh  = (const float*)d_in[12];
  const float* qbih  = (const float*)d_in[13];
  const float* qbhh  = (const float*)d_in[14];
  const float* Wr    = (const float*)d_in[15];
  const float* Ur    = (const float*)d_in[16];
  const float* br    = (const float*)d_in[17];
  const float* Wm    = (const float*)d_in[18];
  const float* Um    = (const float*)d_in[19];
  const float* z1w   = (const float*)d_in[20];
  const float* z1b   = (const float*)d_in[21];
  const float* z2w   = (const float*)d_in[22];
  const float* z2b   = (const float*)d_in[23];
  const float* nmw   = (const float*)d_in[24];
  const float* nmb   = (const float*)d_in[25];
  const float* answ  = (const float*)d_in[26];
  const float* ansb  = (const float*)d_in[27];
  float* out = (float*)d_out;

  char* ws = (char*)d_ws;
  size_t off = 0;
  auto alloc = [&](size_t b) -> char* {
    char* p = ws + off; off = (off + b + 255) & ~(size_t)255; return p;
  };

  // f16 weights
  _Float16* wihf16 = (_Float16*)alloc((size_t)H3 * HD * 2);
  _Float16* whhf16 = (_Float16*)alloc((size_t)H3 * HD * 2);
  _Float16* wihb16 = (_Float16*)alloc((size_t)H3 * HD * 2);
  _Float16* whhb16 = (_Float16*)alloc((size_t)H3 * HD * 2);
  _Float16* qwih16 = (_Float16*)alloc((size_t)H3 * HD * 2);
  _Float16* qwhh16 = (_Float16*)alloc((size_t)H3 * HD * 2);
  _Float16* wrT16  = (_Float16*)alloc((size_t)HD * HD * 2);
  _Float16* urT16  = (_Float16*)alloc((size_t)HD * HD * 2);
  _Float16* wT16   = (_Float16*)alloc((size_t)HD * HD * 2);
  _Float16* uT16   = (_Float16*)alloc((size_t)HD * HD * 2);
  _Float16* z1w16  = (_Float16*)alloc((size_t)HD * H4 * 2);
  _Float16* nmw16  = (_Float16*)alloc((size_t)HD * H3 * 2);
  _Float16* answ16 = (_Float16*)alloc((size_t)VOC * 2 * HD * 2);
  // activations
  _Float16* fin16  = (_Float16*)alloc((size_t)BN * SN * HD * 2);
  _Float16* qemb16 = (_Float16*)alloc((size_t)BN * TQN * HD * 2);
  float* xg_f  = (float*)alloc((size_t)BN * SN * H3 * 4);
  float* xg_b  = (float*)alloc((size_t)BN * SN * H3 * 4);
  float* xg_q  = (float*)alloc((size_t)BN * TQN * H3 * 4);
  float* hs_f  = (float*)alloc((size_t)BN * SN * HD * 4);
  float* hs_b  = (float*)alloc((size_t)BN * SN * HD * 4);
  float* facts32   = (float*)alloc((size_t)BN * SN * HD * 4);
  _Float16* facts16 = (_Float16*)alloc((size_t)BN * SN * HD * 2);
  float* q32   = (float*)alloc((size_t)BN * HD * 4);
  _Float16* q16 = (_Float16*)alloc((size_t)BN * HD * 2);
  _Float16* z16 = (_Float16*)alloc((size_t)BN * SN * H4 * 2);
  float* g1    = (float*)alloc((size_t)BN * SN * HD * 4);
  float* Gbuf  = (float*)alloc((size_t)BN * SN * 4);
  float* A1    = (float*)alloc((size_t)BN * SN * HD * 4);
  float* A2    = (float*)alloc((size_t)BN * SN * HD * 4);
  float* Cbuf  = (float*)alloc((size_t)BN * HD * 4);
  float* Mbuf  = (float*)alloc((size_t)BN * HD * 4);
  _Float16* cat316 = (_Float16*)alloc((size_t)BN * H3 * 2);
  _Float16* mq16   = (_Float16*)alloc((size_t)BN * 2 * HD * 2);

  auto conv = [&](const float* s, _Float16* d, int n) {
    f2h_kernel<<<(n + 255) / 256, 256, 0, stream>>>(s, d, n);
  };
  conv(Wih_f, wihf16, H3 * HD);  conv(Whh_f, whhf16, H3 * HD);
  conv(Wih_b, wihb16, H3 * HD);  conv(Whh_b, whhb16, H3 * HD);
  conv(qWih,  qwih16, H3 * HD);  conv(qWhh,  qwhh16, H3 * HD);
  conv(z1w, z1w16, HD * H4);     conv(nmw, nmw16, HD * H3);
  conv(answ, answ16, VOC * 2 * HD);
  {
    dim3 g(32, 32), b(16, 16);
    t512_f2h_kernel<<<g, b, 0, stream>>>(Wr, wrT16);
    t512_f2h_kernel<<<g, b, 0, stream>>>(Ur, urT16);
    t512_f2h_kernel<<<g, b, 0, stream>>>(Wm, wT16);
    t512_f2h_kernel<<<g, b, 0, stream>>>(Um, uT16);
  }

  // embeddings + positional encoding
  embed_facts_kernel<<<BN * SN, 256, 0, stream>>>(ctx, emb, fin16);
  embed_q_kernel<<<BN * TQN, 256, 0, stream>>>(ques, emb, qemb16);

  // input-side GRU gates (batched over all timesteps)
  wmma_gemm_kernel<<<dim3(H3 / 64, (BN * SN) / 64), 256, 0, stream>>>(
      fin16, wihf16, bih_f, xg_f, BN * SN, H3, HD, 0);
  wmma_gemm_kernel<<<dim3(H3 / 64, (BN * SN) / 64), 256, 0, stream>>>(
      fin16, wihb16, bih_b, xg_b, BN * SN, H3, HD, 0);
  wmma_gemm_kernel<<<dim3(H3 / 64, (BN * TQN) / 64), 256, 0, stream>>>(
      qemb16, qwih16, qbih, xg_q, BN * TQN, H3, HD, 0);

  // three GRU scans in one launch (block 0=fwd, 1=bwd, 2=question)
  gru_scan_kernel<<<3, 1024, SCAN_LDS, stream>>>(
      xg_f, xg_b, xg_q, whhf16, whhb16, qwhh16,
      bhh_f, bhh_b, qbhh, hs_f, hs_b, q32, q16);

  // facts = hf + hb
  {
    int n = BN * SN * HD;
    facts_add_kernel<<<(n + 255) / 256, 256, 0, stream>>>(hs_f, hs_b, facts32, facts16, n);
  }

  // AGRU hoisted terms: A1 = facts@Wr + br ; A2 = facts@W + br
  wmma_gemm_kernel<<<dim3(HD / 64, (BN * SN) / 64), 256, 0, stream>>>(
      facts16, wrT16, br, A1, BN * SN, HD, HD, 0);
  wmma_gemm_kernel<<<dim3(HD / 64, (BN * SN) / 64), 256, 0, stream>>>(
      facts16, wT16, br, A2, BN * SN, HD, HD, 0);

  // M = q
  copyf_kernel<<<(BN * HD + 255) / 256, 256, 0, stream>>>(q32, Mbuf, BN * HD);

  for (int hop = 0; hop < 3; ++hop) {
    build_z_kernel<<<BN * SN, 256, 0, stream>>>(facts32, q32, Mbuf, z16);
    wmma_gemm_kernel<<<dim3(HD / 64, (BN * SN) / 64), 256, 0, stream>>>(
        z16, z1w16, z1b, g1, BN * SN, HD, H4, 1 /*tanh*/);
    attn_kernel<<<BN, SN, 0, stream>>>(g1, z2w, z2b, Gbuf);
    agru_scan_kernel<<<1, 1024, SCAN_LDS, stream>>>(A1, A2, urT16, uT16, Gbuf, Cbuf);
    cat3_kernel<<<BN, 256, 0, stream>>>(Mbuf, Cbuf, q32, cat316);
    wmma_gemm_kernel<<<dim3(HD / 64, BN / 64), 256, 0, stream>>>(
        cat316, nmw16, nmb, Mbuf, BN, HD, H3, 2 /*relu*/);
  }

  // preds = concat(M, q) @ ans_w^T + ans_b
  cat2_kernel<<<BN, 256, 0, stream>>>(Mbuf, q32, mq16);
  wmma_gemm_kernel<<<dim3(VOC / 64, BN / 64), 256, 0, stream>>>(
      mq16, answ16, ansb, out, BN, VOC, 2 * HD, 0);
}
